// HungarianMatcher_55284819034883
// MI455X (gfx1250) — compile-verified, matrix operand order checked by reading, not run on hardware
//
#include <hip/hip_runtime.h>

// Problem constants (from reference)
#define BSZ   16
#define NQ    900
#define NCLS  91
#define NCP   96                 // padded K (multiple of 32)
#define MTGT  1600
#define NTOT  (BSZ*NQ)           // 14400 rows
#define NTILE (NTOT/16)          // 900
#define MTILE (MTGT/16)          // 100
#define W_CLASS 2.0f
#define W_BBOX  5.0f
#define W_GIOU  2.0f
#define F_ALPHA 0.25f

typedef _Float16 half8  __attribute__((ext_vector_type(8)));
typedef _Float16 half16 __attribute__((ext_vector_type(16)));
typedef float    f32x8  __attribute__((ext_vector_type(8)));

union H16 { half16 v; half8 h[2]; };

__device__ __forceinline__ float softplus_f(float x) {
  // log(1+exp(x)) stable == -log_sigmoid(-x)
  float ax = fabsf(x);
  return fmaxf(x, 0.0f) + log1pf(__expf(-ax));
}

// ---------------------------------------------------------------------------
// Kernel 1: focal class-cost table, f32 -> (f16 hi, f16 lo residual), K padded
// tab[n][c] = pos_cost - neg_cost  (gathered later by one-hot WMMA)
// ---------------------------------------------------------------------------
__global__ void k_tab(const float* __restrict__ logits,
                      _Float16* __restrict__ tabH, _Float16* __restrict__ tabL) {
  int i = blockIdx.x * blockDim.x + threadIdx.x;
  if (i >= NTOT * NCP) return;
  int n = i / NCP, c = i % NCP;
  float t = 0.0f;
  if (c < NCLS) {
    float x = logits[n * NCLS + c];
    float p = 1.0f / (1.0f + __expf(-x));
    float pos = F_ALPHA * (1.0f - p) * (1.0f - p) * softplus_f(-x);
    float neg = (1.0f - F_ALPHA) * p * p * softplus_f(x);
    t = pos - neg;
  }
  _Float16 hi = (_Float16)t;
  tabH[i] = hi;
  tabL[i] = (_Float16)(t - (float)hi);
}

// ---------------------------------------------------------------------------
// Kernel 2: box preprocessing: cxcywh -> xyxy + area (queries and targets)
// ---------------------------------------------------------------------------
__global__ void k_boxes(const float4* __restrict__ pboxes, const float4* __restrict__ tboxes,
                        float4* __restrict__ qxy, float* __restrict__ qarea,
                        float4* __restrict__ txy, float* __restrict__ tarea) {
  int i = blockIdx.x * blockDim.x + threadIdx.x;
  if (i < NTOT) {
    float4 b = pboxes[i];
    float4 e;
    e.x = b.x - 0.5f * b.z; e.y = b.y - 0.5f * b.w;
    e.z = b.x + 0.5f * b.z; e.w = b.y + 0.5f * b.w;
    qxy[i] = e;
    qarea[i] = (e.z - e.x) * (e.w - e.y);
  } else if (i < NTOT + MTGT) {
    int m = i - NTOT;
    float4 b = tboxes[m];
    float4 e;
    e.x = b.x - 0.5f * b.z; e.y = b.y - 0.5f * b.w;
    e.z = b.x + 0.5f * b.z; e.w = b.y + 0.5f * b.w;
    txy[m] = e;
    tarea[m] = (e.z - e.x) * (e.w - e.y);
  }
}

// ---------------------------------------------------------------------------
// Kernel 3: one-hot B fragments, pre-swizzled into the 16-bit B-matrix VGPR
// layout (ISA 7.12.2): lane = column (0-15 / 16-31 dup), halves 0-7 -> K=8*hb+j,
// halves 8-15 -> K=16+8*hb+(j-8).  One record of 16 halves per (mtile, s, lane).
// ---------------------------------------------------------------------------
__global__ void k_bfrag(const int* __restrict__ tgt_ids, _Float16* __restrict__ bfrag) {
  int i = blockIdx.x * blockDim.x + threadIdx.x;
  if (i >= MTILE * 3 * 32) return;
  int mt = i / 96;
  int r  = i % 96;
  int s  = r >> 5;          // k-step 0..2
  int lane = r & 31;
  int col = lane & 15, hb = lane >> 4;
  int tid = tgt_ids[mt * 16 + col];
  _Float16* dst = bfrag + (size_t)i * 16;
#pragma unroll
  for (int j = 0; j < 16; ++j) {
    int K  = (j < 8) ? (8 * hb + j) : (16 + 8 * hb + (j - 8));
    int kg = 32 * s + K;
    dst[j] = (kg == tid) ? (_Float16)1.0f : (_Float16)0.0f;
  }
}

// ---------------------------------------------------------------------------
// Kernel 4: main tiled kernel. One wave per 16x16 output tile.
// cost_class via v_wmma_f32_16x16x32_f16 (hi + lo residual, exact to ~2^-21),
// GIoU + L1 + combine on VALU (co-executes with 16-bit WMMA / TRANS pipe).
// ---------------------------------------------------------------------------
__global__ void __launch_bounds__(256) k_main(
    const _Float16* __restrict__ tabH, const _Float16* __restrict__ tabL,
    const _Float16* __restrict__ bfrag,
    const float4* __restrict__ pboxes, const float4* __restrict__ qxy,
    const float*  __restrict__ qarea,
    const float4* __restrict__ tboxes, const float4* __restrict__ txy,
    const float*  __restrict__ tarea,
    float* __restrict__ out) {
  int w = blockIdx.x * (blockDim.x >> 5) + (threadIdx.x >> 5);
  if (w >= NTILE * MTILE) return;           // wave-uniform: EXEC stays all-ones
  int lane = threadIdx.x & 31;
  int nt = w / MTILE, mt = w % MTILE;
  int lo16 = lane & 15, hi = lane >> 4;

  // A-fragment source: row-major hi/lo tables; swizzle = contiguous 8-half runs
  const _Float16* thp = tabH + (size_t)(nt * 16 + lo16) * NCP;
  const _Float16* tlp = tabL + (size_t)(nt * 16 + lo16) * NCP;

  f32x8 acc = {0.f, 0.f, 0.f, 0.f, 0.f, 0.f, 0.f, 0.f};
#pragma unroll
  for (int s = 0; s < 3; ++s) {
    H16 ah, al, bb;
    int ao = 32 * s + 8 * hi;
    ah.h[0] = *(const half8*)(thp + ao);
    ah.h[1] = *(const half8*)(thp + ao + 16);
    al.h[0] = *(const half8*)(tlp + ao);
    al.h[1] = *(const half8*)(tlp + ao + 16);
    const _Float16* b0 = bfrag + ((size_t)(mt * 3 + s) * 32 + lane) * 16;
    bb.h[0] = *(const half8*)(b0);
    bb.h[1] = *(const half8*)(b0 + 8);
    acc = __builtin_amdgcn_wmma_f32_16x16x32_f16(false, ah.v, false, bb.v,
                                                 (short)0, acc, false, false);
    acc = __builtin_amdgcn_wmma_f32_16x16x32_f16(false, al.v, false, bb.v,
                                                 (short)0, acc, false, false);
  }

  // Column (target) data: lane owns column lo16 of the tile
  int tcol = mt * 16 + lo16;
  float4 tb = tboxes[tcol];
  float4 te = txy[tcol];
  float  tA = tarea[tcol];

  int rbase = nt * 16 + 8 * hi;            // D layout: vgpr j -> row 8*hi + j
#pragma unroll
  for (int j = 0; j < 8; ++j) {
    int row = rbase + j;
    float4 qb = pboxes[row];
    float4 qe = qxy[row];
    float  qA = qarea[row];

    // L1 in raw cxcywh
    float l1 = fabsf(qb.x - tb.x) + fabsf(qb.y - tb.y) +
               fabsf(qb.z - tb.z) + fabsf(qb.w - tb.w);

    // GIoU in xyxy
    float ltx = fmaxf(qe.x, te.x), lty = fmaxf(qe.y, te.y);
    float rbx = fminf(qe.z, te.z), rby = fminf(qe.w, te.w);
    float iw = fmaxf(rbx - ltx, 0.0f), ih = fmaxf(rby - lty, 0.0f);
    float inter = iw * ih;
    float uni = qA + tA - inter;
    float iou = inter / uni;
    float ex = fminf(qe.x, te.x), ey = fminf(qe.y, te.y);
    float fx = fmaxf(qe.z, te.z), fy = fmaxf(qe.w, te.w);
    float ew = fmaxf(fx - ex, 0.0f), eh = fmaxf(fy - ey, 0.0f);
    float areaE = ew * eh;
    float giou = iou - (areaE - uni) / areaE;

    float C = W_BBOX * l1 + W_CLASS * acc[j] - W_GIOU * giou;
    if (!(fabsf(C) <= 3.0e38f)) C = 1.0e6f;   // NaN (cmp false) or Inf -> 1e6
    out[(size_t)row * MTGT + tcol] = C;
  }
}

// ---------------------------------------------------------------------------
extern "C" void kernel_launch(void* const* d_in, const int* in_sizes, int n_in,
                              void* d_out, int out_size, void* d_ws, size_t ws_size,
                              hipStream_t stream) {
  (void)in_sizes; (void)n_in; (void)out_size; (void)ws_size;
  const float*  logits  = (const float*)d_in[0];   // [16,900,91]
  const float4* pboxes  = (const float4*)d_in[1];  // [16,900,4]
  const int*    tids    = (const int*)d_in[2];     // [1600]
  const float4* tboxes  = (const float4*)d_in[3];  // [1600,4]
  float* out = (float*)d_out;                      // [16,900,1600]

  // Workspace carve-up (all 16B-aligned section sizes), ~6.2 MB total
  char* ws = (char*)d_ws;
  _Float16* tabH = (_Float16*)ws;  ws += (size_t)NTOT * NCP * 2;   // 2.76 MB
  _Float16* tabL = (_Float16*)ws;  ws += (size_t)NTOT * NCP * 2;   // 2.76 MB
  float4*   qxy  = (float4*)ws;    ws += (size_t)NTOT * 16;        // 230 KB
  float*    qar  = (float*)ws;     ws += (size_t)NTOT * 4;         // 57 KB
  float4*   txy  = (float4*)ws;    ws += (size_t)MTGT * 16;        // 25 KB
  float*    tar  = (float*)ws;     ws += (size_t)MTGT * 4;         // 6 KB
  _Float16* bfr  = (_Float16*)ws;  ws += (size_t)MTILE * 3 * 32 * 16 * 2; // 307 KB

  k_tab  <<<(NTOT * NCP + 255) / 256, 256, 0, stream>>>(logits, tabH, tabL);
  k_boxes<<<(NTOT + MTGT + 255) / 256, 256, 0, stream>>>(pboxes, tboxes, qxy, qar, txy, tar);
  k_bfrag<<<(MTILE * 3 * 32 + 255) / 256, 256, 0, stream>>>(tids, bfr);
  k_main <<<(NTILE * MTILE) / 8, 256, 0, stream>>>(tabH, tabL, bfr,
                                                   pboxes, qxy, qar,
                                                   tboxes, txy, tar, out);
}
